// RGATLayer_8461085573364
// MI455X (gfx1250) — matrix operand order repaired and checked by reading, not compiled, not run on hardware
//
#include <hip/hip_runtime.h>
#include <hip/hip_bf16.h>

typedef __attribute__((ext_vector_type(16))) __bf16 v16bf;
typedef __attribute__((ext_vector_type(8)))  float  v8f;
typedef __attribute__((ext_vector_type(4)))  float  fv4;
typedef __attribute__((ext_vector_type(2)))  float  fv2;

#define N_NODES 50000
#define N_EDGES 800000
#define D 128

// workspace layout (units of 4-byte words)
#define WS_Z      0               // z: N_NODES*D floats = 6,400,000
#define WS_V      6400000         // v[256] + c  (padded)
#define WS_MU     6400384         // segment-max as monotone uint, N_NODES
#define WS_DENOM  6450384         // softmax denom, N_NODES floats
#define WS_EBUF   6500384         // per-edge e / ex, N_EDGES floats

// ---------------- kernel 0: v = w2^T @ W_rel, c = w2 . b_rel ----------------
__global__ void precompute_v(const float* __restrict__ W_rel,
                             const float* __restrict__ b_rel,
                             const float* __restrict__ W_attn,
                             float* __restrict__ ws) {
    int j = threadIdx.x;                 // 0..255
    const float* w2 = W_attn + D;        // attention weights for `end`
    float s = 0.f;
    for (int i = 0; i < D; ++i) s += w2[i] * W_rel[i * (2 * D) + j];
    ws[WS_V + j] = s;
    if (j == 0) {
        float c = 0.f;
        for (int i = 0; i < D; ++i) c += w2[i] * b_rel[i];
        ws[WS_V + 2 * D] = c;
    }
}

// ---------------- kernel 1: zero h / denom / max buffers --------------------
__global__ void init_buffers(float* __restrict__ h, float* __restrict__ ws) {
    int idx = blockIdx.x * blockDim.x + threadIdx.x;
    if (idx < N_NODES * D) h[idx] = 0.f;
    if (idx < N_NODES) {
        ((unsigned*)ws)[WS_MU + idx] = 0u;   // below any mapped finite float
        ws[WS_DENOM + idx] = 0.f;
    }
}

// ---------------- kernel 2: z = x @ W_fc^T via bf16 WMMA --------------------
// one wave per 16x16 output tile; block = 8 waves = full N stripe (128 cols)
__global__ void __launch_bounds__(256) gemm_z(const float* __restrict__ x,
                                              const float* __restrict__ Wfc,
                                              float* __restrict__ ws) {
    float* z = ws + WS_Z;
    const int lane  = threadIdx.x & 31;
    const int wave  = threadIdx.x >> 5;     // tileN: 0..7
    const int tileM = blockIdx.x;           // 0..3124 (50000/16)
    const int half  = lane >> 4;            // lane group 0/1
    const int rowA  = tileM * 16 + (lane & 15);   // x row (M)
    const int rowB  = wave * 16 + (lane & 15);    // W_fc row (N), since B[k][n]=Wfc[n][k]

    v8f acc = {};
    #pragma unroll
    for (int kt = 0; kt < 4; ++kt) {        // K = 128 in steps of 32
        const int k0 = kt * 32;
        v16bf a, b;
        #pragma unroll
        for (int v = 0; v < 8; ++v) {
            // A 16x32 bf16 layout (ISA 7.12.2): lanes0-15 VGPR0-3:K0-7, 4-7:K16-23;
            // lanes16-31 shifted by +8
            int ka = k0 + ((v >> 2) * 16) + half * 8 + (v & 3) * 2;
            // x is read exactly once -> non-temporal, don't pollute L2
            fv2 fa = __builtin_nontemporal_load((const fv2*)(x + rowA * D + ka));
            a[2 * v]     = (__bf16)fa.x;
            a[2 * v + 1] = (__bf16)fa.y;
            // B 32x16 bf16 layout: same K order as A (lanes0-15 K0-7,K16-23; lanes16-31 +8)
            int kb = k0 + ((v >> 2) * 16) + half * 8 + (v & 3) * 2;
            fv2 fb = *(const fv2*)(Wfc + rowB * D + kb);   // tiny, keep cached
            b[2 * v]     = (__bf16)fb.x;
            b[2 * v + 1] = (__bf16)fb.y;
        }
        acc = __builtin_amdgcn_wmma_f32_16x16x32_bf16(
                  false, a, false, b, (short)0, acc, false, false);
    }
    // C/D 16x16 f32 layout: VGPR r -> M = r + half*8, N = lane&15
    const int n = wave * 16 + (lane & 15);
    #pragma unroll
    for (int r = 0; r < 8; ++r) {
        int mm = tileM * 16 + r + half * 8;
        z[mm * D + n] = acc[r];
    }
}

// order-preserving float <-> uint map for atomic max
__device__ __forceinline__ unsigned mapf(float f) {
    unsigned b = __float_as_uint(f);
    return (b & 0x80000000u) ? ~b : (b | 0x80000000u);
}
__device__ __forceinline__ float unmapf(unsigned u) {
    return (u & 0x80000000u) ? __uint_as_float(u ^ 0x80000000u)
                             : __uint_as_float(~u);
}

// ---------------- kernel 3: per-edge attention score + segment max ----------
__global__ void __launch_bounds__(256) edge_score(const float* __restrict__ edge_feat,
                                                  const int* __restrict__ src,
                                                  const int* __restrict__ dst,
                                                  const float* __restrict__ W_attn,
                                                  float* __restrict__ ws) {
    const float* z = ws + WS_Z;
    const float* v = ws + WS_V;
    unsigned* m_u  = (unsigned*)ws + WS_MU;
    float* ebuf    = ws + WS_EBUF;

    int e    = blockIdx.x * 8 + (threadIdx.x >> 5);   // wave per edge
    int lane = threadIdx.x & 31;
    if (e >= N_EDGES) return;
    int s = src[e], d = dst[e];

    fv4 w1 = *(const fv4*)(W_attn + lane * 4);        // attn for z_src
    fv4 ve = *(const fv4*)(v + lane * 4);             // v[0:128]   -> edge_feat
    fv4 vz = *(const fv4*)(v + D + lane * 4);         // v[128:256] -> z_dst
    fv4 zs = *(const fv4*)(z + s * D + lane * 4);     // gathers: keep L2-resident
    fv4 zd = *(const fv4*)(z + d * D + lane * 4);
    // edge_feat is a 410 MB touch-once stream: non-temporal so z stays in L2
    fv4 ef = __builtin_nontemporal_load(
                 (const fv4*)(edge_feat + (long long)e * D + lane * 4));

    float p = w1.x * zs.x + w1.y * zs.y + w1.z * zs.z + w1.w * zs.w
            + ve.x * ef.x + ve.y * ef.y + ve.z * ef.z + ve.w * ef.w
            + vz.x * zd.x + vz.y * zd.y + vz.z * zd.z + vz.w * zd.w;

    #pragma unroll
    for (int off = 16; off; off >>= 1) p += __shfl_xor(p, off, 32);

    if (lane == 0) {
        float a  = p + v[2 * D];                 // + w2.b_rel
        float ev = a > 0.f ? a : 0.01f * a;      // leaky_relu(0.01)
        ebuf[e] = ev;
        atomicMax(&m_u[d], mapf(ev));
    }
}

// ---------------- kernel 4: ex = exp(e - m[dst]); denom += ex ---------------
__global__ void edge_exp(const int* __restrict__ dst, float* __restrict__ ws) {
    int i = blockIdx.x * blockDim.x + threadIdx.x;
    if (i >= N_EDGES) return;
    int d = dst[i];
    float m  = unmapf(((unsigned*)ws)[WS_MU + d]);
    float ex = __expf(ws[WS_EBUF + i] - m);
    ws[WS_EBUF + i] = ex;
    atomicAdd(&ws[WS_DENOM + d], ex);
}

// ---------------- kernel 5: h[dst] += alpha * z[src] ------------------------
__global__ void __launch_bounds__(256) edge_scatter(const int* __restrict__ src,
                                                    const int* __restrict__ dst,
                                                    float* __restrict__ h,
                                                    float* __restrict__ ws) {
    const float* z = ws + WS_Z;
    int e    = blockIdx.x * 8 + (threadIdx.x >> 5);   // wave per edge
    int lane = threadIdx.x & 31;
    if (e >= N_EDGES) return;
    int s = src[e], d = dst[e];
    float alpha = ws[WS_EBUF + e] / ws[WS_DENOM + d];
    fv4 zs = *(const fv4*)(z + s * D + lane * 4);
    float* hp = h + d * D + lane * 4;
    atomicAdd(hp + 0, alpha * zs.x);
    atomicAdd(hp + 1, alpha * zs.y);
    atomicAdd(hp + 2, alpha * zs.z);
    atomicAdd(hp + 3, alpha * zs.w);
}

extern "C" void kernel_launch(void* const* d_in, const int* in_sizes, int n_in,
                              void* d_out, int out_size, void* d_ws, size_t ws_size,
                              hipStream_t stream) {
    const float* x         = (const float*)d_in[0];
    const float* edge_feat = (const float*)d_in[1];
    const int*   src       = (const int*)d_in[2];
    const int*   dst       = (const int*)d_in[3];
    const float* W_fc      = (const float*)d_in[4];
    const float* W_rel     = (const float*)d_in[5];
    const float* b_rel     = (const float*)d_in[6];
    const float* W_attn    = (const float*)d_in[7];
    float* h  = (float*)d_out;
    float* ws = (float*)d_ws;

    precompute_v<<<1, 2 * D, 0, stream>>>(W_rel, b_rel, W_attn, ws);
    init_buffers<<<(N_NODES * D + 255) / 256, 256, 0, stream>>>(h, ws);
    gemm_z<<<N_NODES / 16, 256, 0, stream>>>(x, W_fc, ws);
    edge_score<<<(N_EDGES + 7) / 8, 256, 0, stream>>>(edge_feat, src, dst, W_attn, ws);
    edge_exp<<<(N_EDGES + 255) / 256, 256, 0, stream>>>(dst, ws);
    edge_scatter<<<(N_EDGES + 7) / 8, 256, 0, stream>>>(src, dst, h, ws);
}
